// RegistrationLoss_50654844289743
// MI455X (gfx1250) — matrix operand order repaired and checked by reading, not compiled, not run on hardware
//
#include <hip/hip_runtime.h>
#include <math.h>

// ---------------------------------------------------------------------------
// RegistrationLoss fused kernels for MI455X (gfx1250, wave32, WMMA)
//   B=2, S=D=4096, C=128, CC=64, K=512, TAU=0.1, EPS=0.2
// ---------------------------------------------------------------------------

typedef __attribute__((ext_vector_type(16))) _Float16 v16h;
typedef __attribute__((ext_vector_type(8)))  _Float16 v8h;
typedef __attribute__((ext_vector_type(8)))  float    v8f;

#define INV_TAU 10.0f     // 1/TAU
#define T2      0.04f     // EPS_POSITIVE^2

// ---------------------------------------------------------------------------
// zero a small float region (atomic accumulators)
// ---------------------------------------------------------------------------
__global__ void zero_kernel(float* __restrict__ p, int n) {
  int i = blockIdx.x * blockDim.x + threadIdx.x;
  if (i < n) p[i] = 0.0f;
}

// ---------------------------------------------------------------------------
// pack coords: in [B][3][N] -> out [B][N][4] = (x,y,z, x^2+y^2+z^2)
// ---------------------------------------------------------------------------
__global__ void coord_pack_kernel(const float* __restrict__ coor,
                                  float4* __restrict__ out4,
                                  int Bn, int N) {
  int idx = blockIdx.x * blockDim.x + threadIdx.x;
  int total = Bn * N;
  if (idx >= total) return;
  int b = idx / N, n = idx - b * N;
  const float* base = coor + (size_t)b * 3 * N + n;
  float x = base[0];
  float y = base[(size_t)N];
  float z = base[(size_t)2 * N];
  float4 v; v.x = x; v.y = y; v.z = z; v.w = x * x + y * y + z * z;
  out4[idx] = v;
}

// ---------------------------------------------------------------------------
// L2-normalize features over channel dim and transpose to token-major f16:
//   in  [B][Cdim][N] f32  ->  out [B][N][Cdim] f16
// one wave (32 lanes) per token
// ---------------------------------------------------------------------------
__global__ void normalize_kernel(const float* __restrict__ fea,
                                 _Float16* __restrict__ out,
                                 int Bn, int Cdim, int N) {
  int wave = (blockIdx.x * blockDim.x + threadIdx.x) >> 5;
  int lane = threadIdx.x & 31;
  int total = Bn * N;
  if (wave >= total) return;
  int b = wave / N, n = wave - b * N;
  const float* base = fea + (size_t)b * Cdim * N + n;
  float ss = 0.0f;
  for (int c = lane; c < Cdim; c += 32) {
    float v = base[(size_t)c * N];
    ss += v * v;
  }
  #pragma unroll
  for (int m = 16; m >= 1; m >>= 1) ss += __shfl_xor(ss, m, 32);
  float inv = 1.0f / fmaxf(sqrtf(ss), 1e-12f);
  _Float16* ob = out + ((size_t)b * N + n) * Cdim;
  for (int c = lane; c < Cdim; c += 32)
    ob[c] = (_Float16)(base[(size_t)c * N] * inv);
}

// ---------------------------------------------------------------------------
// offset loss: sum_k ||off_k||  (off: [K][3][1]) -> atomicAdd into *acc
// ---------------------------------------------------------------------------
__global__ void offset_kernel(const float* __restrict__ off,
                              float* __restrict__ acc, int Kn) {
  int k = blockIdx.x * blockDim.x + threadIdx.x;
  float e = 0.0f;
  if (k < Kn) {
    const float* p = off + (size_t)k * 3;
    e = sqrtf(p[0] * p[0] + p[1] * p[1] + p[2] * p[2]);
  }
  #pragma unroll
  for (int m = 16; m >= 1; m >>= 1) e += __shfl_xor(e, m, 32);
  if ((threadIdx.x & 31) == 0) atomicAdd(acc, e);
}

// ---------------------------------------------------------------------------
// Main fused kernel: one wave per 16-row tile; streams all Ncol columns.
// Per 16x16 tile: 4x v_wmma f16 (C=128 cosine), 2x v_wmma (CC=64 cosine),
// VALU 3-D distance; online per-row stats, all branchless (v_cndmask).
// Each iteration loads ALL fragments (12x b128 + coords) into distinct
// registers first -> one load clause + one wait, then the 6-WMMA chain,
// then the stats tail; unroll-by-2 lets the scheduler overlap the next
// iteration's load clause with this iteration's WMMA/VALU tail.
// ---------------------------------------------------------------------------
__global__ __launch_bounds__(32)
void pair_loss_kernel(const _Float16* __restrict__ rowF,
                      const _Float16* __restrict__ rowC,
                      const float4*   __restrict__ rowXYZ,
                      const unsigned char* __restrict__ rowPad,
                      const _Float16* __restrict__ colF,
                      const _Float16* __restrict__ colC,
                      const float4*   __restrict__ colXYZ,
                      int Nrow, int Ncol, float* __restrict__ acc) {
  const int lane = threadIdx.x;
  const int b    = blockIdx.y;
  const int row0 = blockIdx.x << 4;
  const int lhalf = lane >> 4;      // 0 | 1
  const int l15   = lane & 15;
  const int hi0   = lhalf << 3;     // 0 | 8

  // ---- A fragments (rows row0..row0+15), resident for the whole column loop.
  // A 16x32 as loaded here (contiguous k, same as B): lanes 0-15 row=l, K0..15;
  //                        lanes 16-31 row=l-16, K16..31.
  const int arow = row0 + l15;
  const _Float16* aF = rowF + ((size_t)b * Nrow + arow) * 128;
  const _Float16* aC = rowC + ((size_t)b * Nrow + arow) * 64;
  v16h afr[4], acr[2];
  #pragma unroll
  for (int q = 0; q < 4; ++q) {
    v8h lo = *(const v8h*)(aF + 32 * q + (lhalf << 4));
    v8h hi = *(const v8h*)(aF + 32 * q + (lhalf << 4) + 8);
    #pragma unroll
    for (int e = 0; e < 8; ++e) { afr[q][e] = lo[e]; afr[q][e + 8] = hi[e]; }
  }
  #pragma unroll
  for (int q = 0; q < 2; ++q) {
    v8h lo = *(const v8h*)(aC + 32 * q + (lhalf << 4));
    v8h hi = *(const v8h*)(aC + 32 * q + (lhalf << 4) + 8);
    #pragma unroll
    for (int e = 0; e < 8; ++e) { acr[q][e] = lo[e]; acr[q][e + 8] = hi[e]; }
  }

  // Row coordinates for the 8 rows this lane's C/D slots cover (M = hi0+i).
  float4 srow[8];
  #pragma unroll
  for (int i = 0; i < 8; ++i)
    srow[i] = rowXYZ[(size_t)b * Nrow + row0 + hi0 + i];

  float sumf[8], sumc[8], best_dis[8], best_f[8], best_c[8], best_sim[8];
  int   best_idx[8], best_sidx[8];
  #pragma unroll
  for (int i = 0; i < 8; ++i) {
    sumf[i] = 0.0f; sumc[i] = 0.0f;
    best_dis[i] = 3.4e38f; best_f[i] = 0.0f; best_c[i] = 0.0f;
    best_sim[i] = -3.4e38f; best_idx[i] = 0; best_sidx[i] = 0;
  }

  const int ntile = Ncol >> 4;
  const int kbase = lhalf << 4;     // B layout: lanes 0-15 K0..15, 16-31 K16..31
  const _Float16* colFbase = colF + (size_t)b * Ncol * 128 + kbase;
  const _Float16* colCbase = colC + (size_t)b * Ncol * 64 + kbase;
  const float4*   colXbase = colXYZ + (size_t)b * Ncol;

  #pragma unroll 2
  for (int t = 0; t < ntile; ++t) {
    const int cidx = (t << 4) + l15;
    const _Float16* bFp = colFbase + (size_t)cidx * 128;
    const _Float16* bCp = colCbase + (size_t)cidx * 64;

    // Speculative CDNA5 global_prefetch two tiles ahead (past-end prefetches
    // are silently dropped by hardware -> no guard branch).
    __builtin_prefetch((const void*)(bFp + 32 * 128), 0, 1);
    __builtin_prefetch((const void*)(bCp + 32 * 64), 0, 1);

    // ---- load ALL operands for this tile up front (distinct registers) ----
    v16h bfr[4], bcr[2];
    #pragma unroll
    for (int q = 0; q < 4; ++q) bfr[q] = *(const v16h*)(bFp + 32 * q);
    #pragma unroll
    for (int q = 0; q < 2; ++q) bcr[q] = *(const v16h*)(bCp + 32 * q);
    const float4 dcol = colXbase[cidx];

    // ---- 6-WMMA chain (two independent accumulators) ----
    v8f simf = {};
    #pragma unroll
    for (int q = 0; q < 4; ++q)
      simf = __builtin_amdgcn_wmma_f32_16x16x32_f16(false, afr[q], false, bfr[q],
                                                    (short)0, simf, false, false);
    v8f simc = {};
    #pragma unroll
    for (int q = 0; q < 2; ++q)
      simc = __builtin_amdgcn_wmma_f32_16x16x32_f16(false, acr[q], false, bcr[q],
                                                    (short)0, simc, false, false);

    // ---- stats tail (branchless) ----
    #pragma unroll
    for (int i = 0; i < 8; ++i) {
      float dot = srow[i].x * dcol.x + srow[i].y * dcol.y + srow[i].z * dcol.z;
      float dis = srow[i].w + dcol.w - 2.0f * dot;
      float cf = simf[i];
      float cc = simc[i];
      float ef = __expf(cf * INV_TAU - 10.0f);
      float ec = __expf(cc * INV_TAU - 10.0f);
      sumf[i] += ef;
      sumc[i] += (dis > T2) ? ec : 0.0f;          // v_cndmask, no branch
      bool tmin = dis < best_dis[i];              // branchless argmin + captures
      best_idx[i] = tmin ? cidx : best_idx[i];
      best_f[i]   = tmin ? cf   : best_f[i];
      best_c[i]   = tmin ? cc   : best_c[i];
      best_dis[i] = tmin ? dis  : best_dis[i];
      bool tmax = cf > best_sim[i];               // branchless argmax
      best_sidx[i] = tmax ? cidx : best_sidx[i];
      best_sim[i]  = tmax ? cf   : best_sim[i];
    }
  }

  // ---- cross-lane (16-wide) reduction; rows 0..7 live in lanes 0-15,
  //      rows 8..15 in lanes 16-31 (C/D layout).
  #pragma unroll
  for (int i = 0; i < 8; ++i) {
    #pragma unroll
    for (int m = 8; m >= 1; m >>= 1) {
      sumf[i] += __shfl_xor(sumf[i], m, 16);
      sumc[i] += __shfl_xor(sumc[i], m, 16);
      float od = __shfl_xor(best_dis[i], m, 16);
      int   oi = __shfl_xor(best_idx[i], m, 16);
      float of = __shfl_xor(best_f[i],  m, 16);
      float oc = __shfl_xor(best_c[i],  m, 16);
      bool take = (od < best_dis[i]) || ((od == best_dis[i]) && (oi < best_idx[i]));
      best_dis[i] = take ? od : best_dis[i];
      best_idx[i] = take ? oi : best_idx[i];
      best_f[i]   = take ? of : best_f[i];
      best_c[i]   = take ? oc : best_c[i];
      float os  = __shfl_xor(best_sim[i],  m, 16);
      int   osi = __shfl_xor(best_sidx[i], m, 16);
      bool tk2 = (os > best_sim[i]) || ((os == best_sim[i]) && (osi < best_sidx[i]));
      best_sim[i]  = tk2 ? os  : best_sim[i];
      best_sidx[i] = tk2 ? osi : best_sidx[i];
    }
  }

  if (l15 == 0) {
    float negf = 0.0f, negc = 0.0f, cntv = 0.0f, corrv = 0.0f;
    const int rbase = row0 + hi0;
    #pragma unroll
    for (int i = 0; i < 8; ++i) {
      int r = rbase + i;
      bool valid = (rowPad[(size_t)b * Nrow + r] == 0);
      bool corr  = (best_dis[i] <= T2);
      if (valid && corr) {
        // fine: logsumexp over all cols with fixed max bound 10 (=1/TAU, |cos|<=1)
        float lsef = 10.0f + __logf(sumf[i]);
        negf += lsef - best_f[i] * INV_TAU;
        // coarse: neutral cols (dis<=T2, col!=argmin) contribute exp(-1e9)=0;
        // the argmin col (dis<=T2 here) was excluded from sumc -> re-add it.
        float sc = sumc[i] + __expf(best_c[i] * INV_TAU - 10.0f);
        float lsec = 10.0f + __logf(sc);
        negc += lsec - best_c[i] * INV_TAU;
        cntv += 1.0f;
        if (best_sidx[i] == best_idx[i]) corrv += 1.0f;
      }
    }
    atomicAdd(acc + 0, negf);
    atomicAdd(acc + 1, negc);
    atomicAdd(acc + 2, cntv);
    atomicAdd(acc + 3, corrv);
  }
}

// ---------------------------------------------------------------------------
// finalize: combine directional accumulators into the 5 scalar outputs
// acc: [0..3] src-dir {negf,negc,cnt,correct}, [4..7] dst-dir, [8] off_s, [9] off_d
// out: (loss, top1_acc, loss_pairing, loss_coarse, loss_offset)
// ---------------------------------------------------------------------------
__global__ void finalize_kernel(const float* __restrict__ acc,
                                float* __restrict__ out, float Kf) {
  if (threadIdx.x != 0 || blockIdx.x != 0) return;
  float cs = fmaxf(acc[2], 1.0f);
  float cd = fmaxf(acc[6], 1.0f);
  float lp = 0.5f * (acc[0] / cs + acc[4] / cd);
  float lc = 0.5f * (acc[1] / cs + acc[5] / cd);
  float ac = 0.5f * (acc[3] / cs + acc[7] / cd);
  float lo = 0.5f * (acc[8] + acc[9]) / Kf;
  out[0] = lp + lc + lo;
  out[1] = ac;
  out[2] = lp;
  out[3] = lc;
  out[4] = lo;
}

// ---------------------------------------------------------------------------
// launcher
// ---------------------------------------------------------------------------
extern "C" void kernel_launch(void* const* d_in, const int* in_sizes, int n_in,
                              void* d_out, int out_size, void* d_ws, size_t ws_size,
                              hipStream_t stream) {
  (void)in_sizes; (void)n_in; (void)out_size; (void)ws_size;
  constexpr int Bc = 2, Sc = 4096, Dc = 4096, Cc = 128, CCc = 64, Kc = 512;

  const float*         src_coor = (const float*)d_in[0];
  const float*         dst_coor = (const float*)d_in[1];
  const unsigned char* src_pad  = (const unsigned char*)d_in[2];
  const unsigned char* dst_pad  = (const unsigned char*)d_in[3];
  const float*         src_fea  = (const float*)d_in[4];
  const float*         dst_fea  = (const float*)d_in[5];
  const float*         src_cfea = (const float*)d_in[6];
  const float*         dst_cfea = (const float*)d_in[7];
  const float*         src_off  = (const float*)d_in[8];
  const float*         dst_off  = (const float*)d_in[9];

  char* ws = (char*)d_ws;
  size_t off = 0;
  float* acc = (float*)(ws + off);          off += 256;
  float4* sXYZ = (float4*)(ws + off);       off += (size_t)Bc * Sc * 16;
  float4* dXYZ = (float4*)(ws + off);       off += (size_t)Bc * Dc * 16;
  _Float16* sF = (_Float16*)(ws + off);     off += (size_t)Bc * Sc * Cc * 2;
  _Float16* dF = (_Float16*)(ws + off);     off += (size_t)Bc * Dc * Cc * 2;
  _Float16* sC = (_Float16*)(ws + off);     off += (size_t)Bc * Sc * CCc * 2;
  _Float16* dC = (_Float16*)(ws + off);     off += (size_t)Bc * Dc * CCc * 2;

  zero_kernel<<<1, 32, 0, stream>>>(acc, 16);

  {
    int total = Bc * Sc;
    coord_pack_kernel<<<(total + 255) / 256, 256, 0, stream>>>(src_coor, sXYZ, Bc, Sc);
    coord_pack_kernel<<<(total + 255) / 256, 256, 0, stream>>>(dst_coor, dXYZ, Bc, Dc);
  }
  {
    int blocks = (Bc * Sc * 32 + 255) / 256;   // one wave per token
    normalize_kernel<<<blocks, 256, 0, stream>>>(src_fea,  sF, Bc, Cc,  Sc);
    normalize_kernel<<<blocks, 256, 0, stream>>>(dst_fea,  dF, Bc, Cc,  Dc);
    normalize_kernel<<<blocks, 256, 0, stream>>>(src_cfea, sC, Bc, CCc, Sc);
    normalize_kernel<<<blocks, 256, 0, stream>>>(dst_cfea, dC, Bc, CCc, Dc);
  }

  offset_kernel<<<(Kc + 255) / 256, 256, 0, stream>>>(src_off, acc + 8, Kc);
  offset_kernel<<<(Kc + 255) / 256, 256, 0, stream>>>(dst_off, acc + 9, Kc);

  dim3 grid(Sc / 16, Bc);
  // direction src -> dst
  pair_loss_kernel<<<grid, 32, 0, stream>>>(sF, sC, sXYZ, src_pad,
                                            dF, dC, dXYZ, Sc, Dc, acc + 0);
  // direction dst -> src
  pair_loss_kernel<<<grid, 32, 0, stream>>>(dF, dC, dXYZ, dst_pad,
                                            sF, sC, sXYZ, Dc, Sc, acc + 4);

  finalize_kernel<<<1, 1, 0, stream>>>(acc, (float*)d_out, (float)Kc);
}
